// MobGatedDeltaNet_57698590655108
// MI455X (gfx1250) — matrix operand order repaired and checked by reading, not compiled, not run on hardware
//
#include <hip/hip_runtime.h>
#include <hip/hip_bf16.h>
#include <math.h>

typedef __bf16 bf16_t;
typedef __attribute__((ext_vector_type(16))) __bf16 v16bf;
typedef __attribute__((ext_vector_type(8)))  float  v8f;

namespace {
constexpr int kB = 2, kL = 2048, kHID = 1024;
constexpr int kH = 4, kDK = 64, kRATIO = 4, kDV = 128;
constexpr int kKEY = 256, kVAL = 512, kHE = 16, kKS = 4;
constexpr int kM = kB * kL;                                   // 4096 token rows
constexpr int kNALL = kKEY + kKEY + kVAL + kHE + kHE + kVAL;  // 1568 fused proj cols
constexpr int kOffQ = 0, kOffK = 256, kOffV = 512;
constexpr int kOffB = 1024, kOffA = 1040, kOffG = 1056;
}

__device__ __forceinline__ float sigmoidf_(float x) { return 1.f / (1.f + __expf(-x)); }
__device__ __forceinline__ float siluf_(float x) { return x * sigmoidf_(x); }

extern "C" __global__ void cast_f32_bf16_kernel(const float* __restrict__ src,
                                                bf16_t* __restrict__ dst, int n) {
  int i = blockIdx.x * blockDim.x + threadIdx.x;
  if (i < n) dst[i] = (bf16_t)src[i];
}

extern "C" __global__ void zero_f32_kernel(float* __restrict__ p, int n) {
  int i = blockIdx.x * blockDim.x + threadIdx.x;
  if (i < n) p[i] = 0.f;
}

// C[m,n] = sum_k A[m,k] * Bw[n,k].
// 64(M) x 16(N) tile per 128-thread workgroup (4 waves on one WGP).
// A tile 64x32 and B tile 16x32 bf16 staged into double-buffered LDS; each wave
// computes one 16x16 WMMA per K-step from LDS fragments; f32 accumulate.
// Batched over blockIdx.z with element strides (per-head expert expansion).
extern "C" __global__ void gemm_bf16_wmma_kernel(
    const bf16_t* __restrict__ A, int lda, long long strideA,
    const bf16_t* __restrict__ Bw, int ldb, long long strideB,
    float* __restrict__ C, int ldc, long long strideC,
    int M, int N, int K) {
  const int tid = threadIdx.x;   // 0..127
  const int lane = tid & 31;     // wave32
  const int wave = tid >> 5;     // 0..3
  const int tn = blockIdx.x * 16;
  const int tm = blockIdx.y * 64;
  const long long z = blockIdx.z;
  const bf16_t* Ab = A + z * strideA;
  const bf16_t* Bb = Bw + z * strideB;

  __shared__ __align__(16) bf16_t As[2][64][32];
  __shared__ __align__(16) bf16_t Bs[2][16][32];

  auto stage = [&](int k0, int buf) {
    // A: 64 rows x 32 k = 2048 bf16; each thread copies 16 (two 16B transfers)
    {
      int r = tid >> 1;
      int hk = (tid & 1) * 16;
      const bf16_t* src = Ab + (long long)(tm + r) * lda + (k0 + hk);
      float4* dst = (float4*)&As[buf][r][hk];
      dst[0] = ((const float4*)src)[0];
      dst[1] = ((const float4*)src)[1];
      if (k0 + 64 < K) __builtin_prefetch(src + 64, 0, 1);  // -> global_prefetch_b8
    }
    // B: 16 rows x 32 k = 512 bf16; first 64 threads copy 8 each (one 16B transfer)
    if (tid < 64) {
      int r = tid >> 2;
      int qk = (tid & 3) * 8;
      const bf16_t* src = Bb + (long long)(tn + r) * ldb + (k0 + qk);
      *(float4*)&Bs[buf][r][qk] = *(const float4*)src;
      if (k0 + 64 < K) __builtin_prefetch(src + 64, 0, 1);
    }
  };

  // k order used here (contiguous k = 16h + e, not AMD's table order; A and B match):
  //   A 16x32 bf16: lanes 0-15 (M=lane) hold K 0..15; lanes 16-31 hold K 16..31
  //   B 32x16 bf16: lanes 0-15 hold K rows 0..15, lanes 16-31 hold K rows 16..31; N=lane&15
  const int kofA = (lane < 16) ? 0 : 16;
  const int kofB = (lane < 16) ? 0 : 16;
  const int mrow = 16 * wave + (lane & 15);

  v8f acc = {};
  stage(0, 0);
  __syncthreads();
  for (int k0 = 0; k0 < K; k0 += 32) {
    const int buf = (k0 >> 5) & 1;
    if (k0 + 32 < K) stage(k0 + 32, buf ^ 1);
    v16bf a, b;
#pragma unroll
    for (int e = 0; e < 16; ++e) a[e] = As[buf][mrow][kofA + e];
#pragma unroll
    for (int e = 0; e < 16; ++e) b[e] = Bs[buf][lane & 15][kofB + e];
    acc = __builtin_amdgcn_wmma_f32_16x16x32_bf16(false, a, false, b, (short)0, acc,
                                                  false, false);
    __syncthreads();
  }

  // C/D layout: VGPR j holds row (j + 8*(lane>=16)), col = lane&15
  float* Crow = C + z * strideC;
  const int rbase = tm + 16 * wave + ((lane >> 4) * 8);
  const int col = tn + (lane & 15);
#pragma unroll
  for (int j = 0; j < 8; ++j) Crow[(long long)(rbase + j) * ldc + col] = acc[j];
}

// Causal depthwise conv (KS=4) + SiLU, reading a column slice of fused Y.
extern "C" __global__ void dwconv_silu_kernel(const float* __restrict__ src, int ld,
                                              const float* __restrict__ w, int C,
                                              float* __restrict__ dst,
                                              bf16_t* __restrict__ dstbf) {
  int i = blockIdx.x * blockDim.x + threadIdx.x;
  if (i >= kM * C) return;
  int m = i / C, c = i % C;
  int b = m / kL, l = m % kL;
  float acc = 0.f;
#pragma unroll
  for (int t = 0; t < kKS; ++t) {
    int lp = l - (kKS - 1) + t;
    if (lp >= 0) acc += src[(long long)(b * kL + lp) * ld + c] * w[c * kKS + t];
  }
  float y = siluf_(acc);
  dst[i] = y;
  if (dstbf) dstbf[i] = (bf16_t)y;
}

// Routing: softmax over 3 expert logits, top-2, shared expert, beta & decay g.
extern "C" __global__ void route_kernel(const float* __restrict__ qc,
                                        const float* __restrict__ Y,
                                        const float* __restrict__ Wgate,
                                        const float* __restrict__ A_log,
                                        const float* __restrict__ dt_bias,
                                        float* __restrict__ wr,
                                        float* __restrict__ beta,
                                        float* __restrict__ gdec) {
  int i = blockIdx.x * blockDim.x + threadIdx.x;
  if (i >= kM * kH) return;
  int m = i / kH, h = i % kH;
  const float* q = qc + (long long)m * kKEY + h * kDK;
  float lg[3];
#pragma unroll
  for (int n = 0; n < 3; ++n) {
    float s = 0.f;
    for (int d = 0; d < kDK; ++d) s += q[d] * Wgate[n * kDK + d];
    lg[n] = s;
  }
  float mx = fmaxf(lg[0], fmaxf(lg[1], lg[2]));
  float ex[3], ssum = 0.f;
#pragma unroll
  for (int n = 0; n < 3; ++n) { ex[n] = __expf(lg[n] - mx); ssum += ex[n]; }
#pragma unroll
  for (int n = 0; n < 3; ++n) ex[n] /= ssum;
  int i0 = 0;
  if (ex[1] > ex[i0]) i0 = 1;
  if (ex[2] > ex[i0]) i0 = 2;
  int i1 = (i0 == 0) ? 1 : 0;
#pragma unroll
  for (int n = 0; n < 3; ++n)
    if (n != i0 && ex[n] > ex[i1]) i1 = n;
  float t0 = ex[i0], t1 = ex[i1], ts = t0 + t1;
  t0 /= ts; t1 /= ts;
  float full[kRATIO] = {1.f, 0.f, 0.f, 0.f};  // shared expert weight 1/SHARED
  full[1 + i0] += t0;
  full[1 + i1] += t1;
  // sum(full) == 2 -> w_r = full * 0.5
#pragma unroll
  for (int r = 0; r < kRATIO; ++r) {
    int he = h * kRATIO + r;
    float w = full[r] * 0.5f;
    wr[m * kHE + he] = w;
    float mask = (full[r] > 0.f) ? 1.f : 0.f;
    float bl = Y[(long long)m * kNALL + kOffB + he];
    beta[m * kHE + he] = sigmoidf_(bl) * mask;
    float al = Y[(long long)m * kNALL + kOffA + he] + dt_bias[he];
    float sp = (al > 20.f) ? al : log1pf(__expf(al));
    gdec[m * kHE + he] = -__expf(A_log[he]) * sp * mask;
  }
}

extern "C" __global__ void l2norm_kernel(float* __restrict__ buf, float post_scale) {
  int i = blockIdx.x * blockDim.x + threadIdx.x;
  if (i >= kM * kHE) return;
  float* p = buf + (long long)i * kDK;
  float ss = 0.f;
#pragma unroll 8
  for (int d = 0; d < kDK; ++d) ss += p[d] * p[d];
  float inv = rsqrtf(ss + 1e-6f) * post_scale;
#pragma unroll 8
  for (int d = 0; d < kDK; ++d) p[d] *= inv;
}

// Gated delta-rule recurrence. One workgroup per (batch, head-expert) memory;
// thread c owns state column S[:, c] in VGPRs; k/q broadcast via LDS (double-buffered,
// one barrier per step). MoE combine fused via f32 global atomics.
extern "C" __global__ void delta_recurrence_kernel(
    const float* __restrict__ qe, const float* __restrict__ ke,
    const float* __restrict__ vc, const float* __restrict__ beta,
    const float* __restrict__ gdec, const float* __restrict__ wr,
    float* __restrict__ omix) {
  const int bhe = blockIdx.x;
  const int b = bhe / kHE, he = bhe % kHE, h = he / kRATIO;
  const int c = threadIdx.x;  // 0..127 = DV column
  float S[kDK];
#pragma unroll
  for (int d = 0; d < kDK; ++d) S[d] = 0.f;
  __shared__ float kq[2][2 * kDK];
  for (int l = 0; l < kL; ++l) {
    const int m = b * kL + l;
    const int buf = l & 1;
    if (c < kDK)
      kq[buf][c] = ke[(long long)m * (kHE * kDK) + he * kDK + c];
    else
      kq[buf][c] = qe[(long long)m * (kHE * kDK) + he * kDK + (c - kDK)];
    __syncthreads();
    const float gt = gdec[m * kHE + he];
    const float bt = beta[m * kHE + he];
    const float w = wr[m * kHE + he];
    const float eg = __expf(gt);
    const float* kk = &kq[buf][0];
    const float* qq = &kq[buf][kDK];
    float dot = 0.f;
#pragma unroll
    for (int d = 0; d < kDK; ++d) dot += kk[d] * S[d];
    const float vt = vc[(long long)m * kVAL + h * kDV + c];
    const float u = (vt - eg * dot) * bt;  // (v - k^T(S*eg)) * beta
    float o = 0.f;
#pragma unroll
    for (int d = 0; d < kDK; ++d) {
      S[d] = S[d] * eg + kk[d] * u;
      o += qq[d] * S[d];
    }
    if (w != 0.f)
      atomicAdd(&omix[(long long)m * kVAL + h * kDV + c], w * o);
  }
}

// FusedRMSNormSwishGate, writes bf16 operand for the output projection GEMM.
extern "C" __global__ void finalize_kernel(const float* __restrict__ omix,
                                           const float* __restrict__ Y,
                                           const float* __restrict__ norm_w,
                                           bf16_t* __restrict__ omix_bf) {
  int i = blockIdx.x * blockDim.x + threadIdx.x;
  if (i >= kM * kH) return;
  int m = i / kH, h = i % kH;
  const float* op = omix + (long long)m * kVAL + h * kDV;
  float ss = 0.f;
#pragma unroll 8
  for (int c = 0; c < kDV; ++c) ss += op[c] * op[c];
  float inv = rsqrtf(ss * (1.f / kDV) + 1e-5f);
  bf16_t* ob = omix_bf + (long long)m * kVAL + h * kDV;
#pragma unroll 8
  for (int c = 0; c < kDV; ++c) {
    float gate = Y[(long long)m * kNALL + kOffG + h * kDV + c];
    ob[c] = (bf16_t)(op[c] * inv * norm_w[c] * siluf_(gate));
  }
}

extern "C" void kernel_launch(void* const* d_in, const int* in_sizes, int n_in,
                              void* d_out, int out_size, void* d_ws, size_t ws_size,
                              hipStream_t stream) {
  const float* x       = (const float*)d_in[0];
  const float* Wq      = (const float*)d_in[1];
  const float* Wk      = (const float*)d_in[2];
  const float* Wv      = (const float*)d_in[3];
  const float* Wb      = (const float*)d_in[4];
  const float* Wa      = (const float*)d_in[5];
  const float* Wg      = (const float*)d_in[6];
  const float* Wo      = (const float*)d_in[7];
  const float* conv_q  = (const float*)d_in[8];
  const float* conv_k  = (const float*)d_in[9];
  const float* conv_v  = (const float*)d_in[10];
  const float* Wq_exp  = (const float*)d_in[11];
  const float* Wk_exp  = (const float*)d_in[12];
  const float* W_gate  = (const float*)d_in[13];
  const float* A_log   = (const float*)d_in[14];
  const float* dt_bias = (const float*)d_in[15];
  const float* norm_w  = (const float*)d_in[16];
  float* out = (float*)d_out;

  char* wsp = (char*)d_ws;
  auto alloc = [&](size_t bytes) -> char* {
    char* p = wsp;
    wsp += (bytes + 255) & ~(size_t)255;
    return p;
  };
  bf16_t* xbf    = (bf16_t*)alloc((size_t)kM * kHID * 2);
  bf16_t* Wallbf = (bf16_t*)alloc((size_t)kNALL * kHID * 2);
  bf16_t* Wqe_bf = (bf16_t*)alloc((size_t)kH * kRATIO * kDK * kDK * 2);
  bf16_t* Wke_bf = (bf16_t*)alloc((size_t)kH * kRATIO * kDK * kDK * 2);
  bf16_t* Wo_bf  = (bf16_t*)alloc((size_t)kHID * kVAL * 2);
  float*  Yb     = (float*)alloc((size_t)kM * kNALL * 4);
  float*  qcv    = (float*)alloc((size_t)kM * kKEY * 4);
  float*  kcv    = (float*)alloc((size_t)kM * kKEY * 4);
  float*  vcv    = (float*)alloc((size_t)kM * kVAL * 4);
  bf16_t* qcbf   = (bf16_t*)alloc((size_t)kM * kKEY * 2);
  bf16_t* kcbf   = (bf16_t*)alloc((size_t)kM * kKEY * 2);
  float*  qe     = (float*)alloc((size_t)kM * kHE * kDK * 4);
  float*  ke     = (float*)alloc((size_t)kM * kHE * kDK * 4);
  float*  wrb    = (float*)alloc((size_t)kM * kHE * 4);
  float*  betab  = (float*)alloc((size_t)kM * kHE * 4);
  float*  gdecb  = (float*)alloc((size_t)kM * kHE * 4);
  float*  omix   = (float*)alloc((size_t)kM * kVAL * 4);
  bf16_t* omixbf = (bf16_t*)alloc((size_t)kM * kVAL * 2);

  auto cast = [&](const float* s, bf16_t* d, int n) {
    cast_f32_bf16_kernel<<<(n + 255) / 256, 256, 0, stream>>>(s, d, n);
  };
  cast(x, xbf, kM * kHID);
  cast(Wq, Wallbf + (size_t)kOffQ * kHID, kKEY * kHID);
  cast(Wk, Wallbf + (size_t)kOffK * kHID, kKEY * kHID);
  cast(Wv, Wallbf + (size_t)kOffV * kHID, kVAL * kHID);
  cast(Wb, Wallbf + (size_t)kOffB * kHID, kHE * kHID);
  cast(Wa, Wallbf + (size_t)kOffA * kHID, kHE * kHID);
  cast(Wg, Wallbf + (size_t)kOffG * kHID, kVAL * kHID);
  cast(Wq_exp, Wqe_bf, kH * kRATIO * kDK * kDK);
  cast(Wk_exp, Wke_bf, kH * kRATIO * kDK * kDK);
  cast(Wo, Wo_bf, kHID * kVAL);

  // Fused projections: Y(4096x1568) = x(4096x1024) @ Wall^T
  gemm_bf16_wmma_kernel<<<dim3(kNALL / 16, kM / 64, 1), 128, 0, stream>>>(
      xbf, kHID, 0, Wallbf, kHID, 0, Yb, kNALL, 0, kM, kNALL, kHID);

  dwconv_silu_kernel<<<(kM * kKEY + 255) / 256, 256, 0, stream>>>(
      Yb + kOffQ, kNALL, conv_q, kKEY, qcv, qcbf);
  dwconv_silu_kernel<<<(kM * kKEY + 255) / 256, 256, 0, stream>>>(
      Yb + kOffK, kNALL, conv_k, kKEY, kcv, kcbf);
  dwconv_silu_kernel<<<(kM * kVAL + 255) / 256, 256, 0, stream>>>(
      Yb + kOffV, kNALL, conv_v, kVAL, vcv, (bf16_t*)nullptr);

  route_kernel<<<(kM * kH + 255) / 256, 256, 0, stream>>>(
      qcv, Yb, W_gate, A_log, dt_bias, wrb, betab, gdecb);

  // Per-head expert expansion (batched over H): (4096x64) @ (64x256)
  gemm_bf16_wmma_kernel<<<dim3((kRATIO * kDK) / 16, kM / 64, kH), 128, 0, stream>>>(
      qcbf, kKEY, (long long)kDK, Wqe_bf, kDK, (long long)kRATIO * kDK * kDK,
      qe, kHE * kDK, (long long)kRATIO * kDK, kM, kRATIO * kDK, kDK);
  gemm_bf16_wmma_kernel<<<dim3((kRATIO * kDK) / 16, kM / 64, kH), 128, 0, stream>>>(
      kcbf, kKEY, (long long)kDK, Wke_bf, kDK, (long long)kRATIO * kDK * kDK,
      ke, kHE * kDK, (long long)kRATIO * kDK, kM, kRATIO * kDK, kDK);

  l2norm_kernel<<<(kM * kHE + 255) / 256, 256, 0, stream>>>(qe, 0.125f);  // * DK^-0.5
  l2norm_kernel<<<(kM * kHE + 255) / 256, 256, 0, stream>>>(ke, 1.0f);

  zero_f32_kernel<<<(kM * kVAL + 255) / 256, 256, 0, stream>>>(omix, kM * kVAL);

  delta_recurrence_kernel<<<kB * kHE, kDV, 0, stream>>>(
      qe, ke, vcv, betab, gdecb, wrb, omix);

  finalize_kernel<<<(kM * kH + 255) / 256, 256, 0, stream>>>(omix, Yb, norm_w, omixbf);

  // Output projection: out(4096x1024) = o(4096x512) @ Wo^T
  gemm_bf16_wmma_kernel<<<dim3(kHID / 16, kM / 64, 1), 128, 0, stream>>>(
      omixbf, kVAL, 0, Wo_bf, kVAL, 0, out, kHID, 0, kM, kHID, kVAL);
}
